// SubOut_60206851555968
// MI455X (gfx1250) — matrix operand order repaired and checked by reading, not compiled, not run on hardware
//
#include <hip/hip_runtime.h>

// ---------------------------------------------------------------------------
// CDNA5 (gfx1250) fused OuterProductUpdate
//   proj:  u_p = u_enc@W + b, e_p = e_enc@W + b      (bf16 WMMA, K=D=256)
//   pair:  op = u_p @ e_p^T, mask = ue*sigmoid(...)  (bf16 WMMA, fused gate,
//          A-tile staged into LDS via TDM tensor_load_to_lds)
//          rowsum[b,u] = sum_e mask, colsum[b,e] = sum_u mask (atomics)
//   final: out = [rowsum@u_enc , colsum@e_enc]
// ---------------------------------------------------------------------------

typedef __attribute__((ext_vector_type(16))) __bf16  v16bf;
typedef __attribute__((ext_vector_type(8)))  __bf16  v8bf;
typedef __attribute__((ext_vector_type(8)))  float   v8f;
typedef __attribute__((ext_vector_type(4)))  unsigned v4u;
typedef __attribute__((ext_vector_type(8)))  unsigned v8u;

union Frag {
    v16bf v;
    v8bf  h[2];
    uint4 q[2];
};

// Sizes
#define B_  8
#define U_  1024
#define E_  1024
#define D_  256
#define P_  256

// ---------------------------------------------------------------------------
// Kernel 0: zero the rowsum/colsum accumulators (16384 floats)
// ---------------------------------------------------------------------------
__global__ void zero_kernel(float* __restrict__ p, int n) {
    int g = blockIdx.x * blockDim.x + threadIdx.x;
    if (g < n) p[g] = 0.0f;
}

// ---------------------------------------------------------------------------
// Kernel 1: Wt[p][d] = bf16(W[d][p])   (256x256)  -- hardware v_cvt to bf16
// ---------------------------------------------------------------------------
__global__ void wt_kernel(const float* __restrict__ W, __bf16* __restrict__ Wt) {
    int p = blockIdx.x;   // 0..255
    int d = threadIdx.x;  // 0..255
    Wt[p * D_ + d] = (__bf16)W[d * P_ + p];
}

// ---------------------------------------------------------------------------
// Kernel 2: projection GEMM. 16384 rows (first 8192 = u, rest = e), K=256,
// N=256. One block = 16-row strip, 4 waves; wave handles 4 N-tiles of 16.
// ---------------------------------------------------------------------------
__global__ void __launch_bounds__(128)
proj_kernel(const float* __restrict__ u_enc,
            const float* __restrict__ e_enc,
            const float* __restrict__ w_bias,
            const __bf16* __restrict__ Wt,
            __bf16* __restrict__ up,
            __bf16* __restrict__ ep) {
    const int tid  = threadIdx.x;
    const int wave = tid >> 5;
    const int lane = tid & 31;
    const int h    = lane >> 4;   // lane half: selects K sub-range
    const int m    = lane & 15;   // A-row / B-col within tile

    const long row0 = (long)blockIdx.x * 16;
    const bool is_u = (row0 < (long)B_ * U_);
    const float* src = is_u ? (u_enc + row0 * D_) : (e_enc + (row0 - (long)B_ * U_) * D_);
    __bf16* dst = is_u ? (up + row0 * P_) : (ep + (row0 - (long)B_ * U_) * P_);

    // A fragments: row (row0+m), all of K=256 as 8 ksteps of 32.
    // Layout (16-bit A 16x32): lane m+16h holds K = ks*32 + {8h..8h+7, 16+8h..16+8h+7}
    Frag a[8];
    const float* arow = src + (long)m * D_;
#pragma unroll
    for (int ks = 0; ks < 8; ++ks) {
        const float* p0 = arow + ks * 32 + 8 * h;           // 32B aligned
        v8f flo = *(const v8f*)(p0);
        v8f fhi = *(const v8f*)(p0 + 16);
        a[ks].h[0] = __builtin_convertvector(flo, v8bf);    // v_cvt_pk_bf16_f32
        a[ks].h[1] = __builtin_convertvector(fhi, v8bf);
    }

    for (int t = 0; t < 4; ++t) {
        const int p0 = (wave * 4 + t) * 16;  // N-tile base column
        v8f acc = {};
        // B layout (16-bit B 32x16): lane n+16h holds K = ks*32 + {8h..8h+7, 16+8h..16+8h+7} of col n.
        const __bf16* brow = Wt + (long)(p0 + m) * D_;
#pragma unroll
        for (int ks = 0; ks < 8; ++ks) {
            Frag bf;
            const __bf16* q = brow + ks * 32 + 8 * h;
            bf.q[0] = *(const uint4*)(q);
            bf.q[1] = *(const uint4*)(q + 16);
            acc = __builtin_amdgcn_wmma_f32_16x16x32_bf16(
                false, a[ks].v, false, bf.v, (short)0, acc, false, false);
        }
        const float bias = w_bias[p0 + m];
        // C layout: VGPR v, lanes 0-15 -> row v; lanes 16-31 -> row v+8; col = m.
#pragma unroll
        for (int v = 0; v < 8; ++v) {
            dst[(long)(v + 8 * h) * P_ + p0 + m] = (__bf16)(acc[v] + bias);
        }
    }
}

// ---------------------------------------------------------------------------
// Kernel 3: fused pair GEMM + sigmoid gate + row/col reductions.
// Grid: 4096 blocks x 256 threads (8 waves). Block -> (b, u-tile, 8 e-tiles);
// wave -> one 16x16 tile: 8x v_wmma_f32_16x16x32_bf16 over K=256.
// The shared 16x256 bf16 u_p tile (8 KB) is DMA'd into LDS once per block by
// the Tensor Data Mover; all 8 waves read A-fragments via ds_load_b128.
// ---------------------------------------------------------------------------
__global__ void __launch_bounds__(256)
pair_kernel(const float* __restrict__ pair_enc,
            const float* __restrict__ ue_mask,
            const float* __restrict__ bpp_w,
            const float* __restrict__ bpp_b,
            const __bf16* __restrict__ up,
            const __bf16* __restrict__ ep,
            float* __restrict__ rowsum,
            float* __restrict__ colsum) {
    __shared__ __align__(16) __bf16 atile[16 * 256];   // 8 KB

    const int tid  = threadIdx.x;
    const int wave = tid >> 5;
    const int lane = tid & 31;
    const int h    = lane >> 4;
    const int m    = lane & 15;

    const int bx = blockIdx.x;            // 4096
    const int b  = bx >> 9;               // 8 batches
    const int ut = (bx >> 3) & 63;        // 64 u-tiles
    const int et = (bx & 7) * 8 + wave;   // 64 e-tiles, 8 per block
    const int u0 = ut * 16;
    const int e0 = et * 16;

    // LDS byte offset of atile: flat aperture keeps the LDS offset in the low
    // 32 bits. The ptrtoint also makes `atile` escape into the asm below, so
    // the TDM "memory" clobber is known to write it (keeps ds loads alive).
    const unsigned lds_base = (unsigned)(size_t)(&atile[0]);

    // ---- TDM: DMA u_p[b, u0:u0+16, 0:256] (bf16, row stride 256) -> LDS ----
    if (tid < 32) {  // wave 0 issues the tensor DMA (EXEC-independent)
        unsigned long long ga =
            (unsigned long long)(size_t)(up + (size_t)(b * U_ + u0) * P_);
        v4u g0;
        g0[0] = 1u;                                   // count=1, user mode
        g0[1] = lds_base;                             // lds_addr = &atile
        g0[2] = (unsigned)(ga & 0xFFFFFFFFu);         // global_addr[31:0]
        g0[3] = (unsigned)((ga >> 32) & 0x01FFFFFFu)  // global_addr[56:32]
              | (2u << 30);                           // type = 2 (image)
        v8u g1;
        g1[0] = 0x00010000u;   // wg_mask=0, data_size=1 (2 bytes)
        g1[1] = (unsigned)(D_ & 0xFFFF) << 16;        // tensor_dim0 = 256 (lo16)
        g1[2] = ((unsigned)16 << 16);                 // tensor_dim0 hi=0, tensor_dim1 = 16
        g1[3] = ((unsigned)D_ << 16);                 // tensor_dim1 hi=0, tile_dim0 = 256
        g1[4] = 16u;                                  // tile_dim1 = 16, tile_dim2 = 0
        g1[5] = (unsigned)D_;                         // tensor_dim0_stride = 256
        g1[6] = 0u;                                   // stride hi, dim1_stride lo
        g1[7] = 0u;
        asm volatile("tensor_load_to_lds %0, %1"
                     :: "s"(g0), "s"(g1)
                     : "memory");
        __builtin_amdgcn_s_wait_tensorcnt(0);
    }
    __syncthreads();

    const __bf16* brow = ep + ((long)(b * E_ + e0 + m)) * P_;

    v8f acc = {};
#pragma unroll
    for (int ks = 0; ks < 8; ++ks) {
        Frag a, bf;
        // A from LDS (16-bit A 16x32 layout): lane m+16h holds
        //   K = ks*32 + {8h..8h+7, 16+8h..16+8h+7}
        const __bf16* pa = atile + m * D_ + ks * 32 + 8 * h;
        a.q[0] = *(const uint4*)(pa);        // ds_load_b128
        a.q[1] = *(const uint4*)(pa + 16);
        const __bf16* pb = brow + ks * 32 + 8 * h;
        bf.q[0] = *(const uint4*)(pb);       // global_load_b128 (L2-resident)
        bf.q[1] = *(const uint4*)(pb + 16);
        acc = __builtin_amdgcn_wmma_f32_16x16x32_bf16(
            false, a.v, false, bf.v, (short)0, acc, false, false);
    }

    const float bw  = *bpp_w;
    const float bb2 = *bpp_b;

    float csum = 0.0f;
    const long base = ((long)(b * U_ + u0)) * E_ + e0 + m;  // col = e0+m
#pragma unroll
    for (int v = 0; v < 8; ++v) {
        const long idx = base + (long)(v + 8 * h) * E_;     // row = u0+v+8h
        const float pe = pair_enc[idx];
        const float qm = ue_mask[idx];
        const float opv = qm * acc[v];
        const float x = bw * pe + bb2 + opv;
        const float sg = 1.0f / (1.0f + __expf(-x));
        const float mk = qm * sg;
        csum += mk;
        // row-reduce over the 16 e-columns (butterfly within the lane-half)
        float r = mk;
        r += __shfl_xor(r, 1);
        r += __shfl_xor(r, 2);
        r += __shfl_xor(r, 4);
        r += __shfl_xor(r, 8);
        if (m == 0) atomicAdd(&rowsum[b * U_ + u0 + v + 8 * h], r);
    }
    // col-reduce: per-lane sum over 8 rows, then combine the two halves
    csum += __shfl_xor(csum, 16);
    if (lane < 16) atomicAdd(&colsum[b * E_ + e0 + m], csum);
}

// ---------------------------------------------------------------------------
// Kernel 4: out[b, 0:256]   = sum_u rowsum[b,u] * u_enc[b,u,:]
//           out[b, 256:512] = sum_e colsum[b,e] * e_enc[b,e,:]
// ---------------------------------------------------------------------------
__global__ void __launch_bounds__(256)
finalize_kernel(const float* __restrict__ u_enc,
                const float* __restrict__ e_enc,
                const float* __restrict__ rowsum,
                const float* __restrict__ colsum,
                float* __restrict__ out) {
    const int g = blockIdx.x * blockDim.x + threadIdx.x;  // 0..4095
    const int b = g >> 9;
    const int r = g & 511;
    const bool is_u = (r < D_);
    const int d = is_u ? r : r - D_;
    const float* enc = is_u ? u_enc : e_enc;
    const float* w   = is_u ? rowsum : colsum;
    const long rowbase = (long)b * U_;
    float acc = 0.0f;
#pragma unroll 4
    for (int i = 0; i < U_; ++i) {
        acc = fmaf(w[rowbase + i], enc[(rowbase + i) * D_ + d], acc);
    }
    out[g] = acc;
}

// ---------------------------------------------------------------------------
// Launch
// ---------------------------------------------------------------------------
extern "C" void kernel_launch(void* const* d_in, const int* in_sizes, int n_in,
                              void* d_out, int out_size, void* d_ws, size_t ws_size,
                              hipStream_t stream) {
    const float* u_enc    = (const float*)d_in[0];
    const float* e_enc    = (const float*)d_in[1];
    const float* pair_enc = (const float*)d_in[2];
    const float* ue_mask  = (const float*)d_in[3];
    const float* w_kernel = (const float*)d_in[4];
    const float* w_bias   = (const float*)d_in[5];
    const float* bpp_w    = (const float*)d_in[6];
    const float* bpp_b    = (const float*)d_in[7];
    float* out = (float*)d_out;

    // Workspace layout (total ~8.6 MB):
    char* ws = (char*)d_ws;
    __bf16* Wt = (__bf16*)ws;                                   // 256*256*2   = 128 KB
    __bf16* up = (__bf16*)(ws + (size_t)131072);                // 8192*256*2  = 4 MB
    __bf16* ep = (__bf16*)(ws + (size_t)131072 + 4194304);
    float* rowsum = (float*)(ws + (size_t)131072 + 2ull * 4194304);  // 8192 f32
    float* colsum = rowsum + (size_t)B_ * U_;                        // 8192 f32

    zero_kernel<<<64, 256, 0, stream>>>(rowsum, 2 * B_ * U_);
    wt_kernel<<<P_, D_, 0, stream>>>(w_kernel, Wt);
    proj_kernel<<<(B_ * U_ + B_ * E_) / 16, 128, 0, stream>>>(u_enc, e_enc, w_bias, Wt, up, ep);
    pair_kernel<<<4096, 256, 0, stream>>>(pair_enc, ue_mask, bpp_w, bpp_b, up, ep, rowsum, colsum);
    finalize_kernel<<<16, 256, 0, stream>>>(u_enc, e_enc, rowsum, colsum, out);
}
